// ESN_31945966747889
// MI455X (gfx1250) — compile-verified, matrix operand order checked by reading, not run on hardware
//
#include <hip/hip_runtime.h>
#include <hip/hip_bf16.h>

#define HIDDEN   200
#define T_STEPS  262144
#define WASHOUT  200
#define NCHUNK   7        // K padded to 7*32 = 224 (covers 200 state + u-channel)
#define ROWS     256      // M padded to 16 tiles, 1 per wave, 16 waves
#define NWAVES   16
#define NPART    32       // per lane-group partials (2 per wave)
#define U_SLOT   HIDDEN   // x-slot 200 carries u_t (augmented input channel)

typedef __attribute__((ext_vector_type(16))) _Float16 v16h;
typedef __attribute__((ext_vector_type(8)))  _Float16 v8h;
typedef __attribute__((ext_vector_type(8)))  float    v8f;

__device__ __forceinline__ float fast_tanh(float x) {
#if __has_builtin(__builtin_amdgcn_tanhf)
  return __builtin_amdgcn_tanhf(x);          // v_tanh_f32 (gfx1250 trans op)
#elif __has_builtin(__builtin_amdgcn_tanh_f32)
  return __builtin_amdgcn_tanh_f32(x);
#else
  return tanhf(x);
#endif
}

// Persistent single-workgroup ESN scan, 16 wave32s, one 16-row WMMA tile each.
//  - Augmented weights [W | w_in] (f16, 256x224) resident in A-operand VGPRs.
//  - Augmented state [x ; u_t] f16 in LDS, ping-pong; B = state replicated over N.
//  - 7-deep v_wmma_f32_16x16x32_f16 chain gives y = W x + w_in u_t directly.
//  - readout: per lane-group plain ds_store (no shuffle, no atomic); the 32-way
//    sum + global store run between s_barrier_signal and s_barrier_wait.
__global__ __launch_bounds__(512, 1)
void esn_scan_kernel(const float* __restrict__ u,
                     const float* __restrict__ w_res,
                     const float* __restrict__ w_in,
                     const float* __restrict__ w_out,
                     float* __restrict__ out) {
  __shared__ __align__(16) _Float16 xh[2][ROWS];     // f16 state, ping-pong
  __shared__ __align__(16) float partials[2][NPART]; // per-group readout, parity

  const int tid  = threadIdx.x;
  const int wv   = tid >> 5;              // wave id 0..15
  const int lane = tid & 31;
  const int g    = (lane >= 16) ? 8 : 0;  // K-group offset (A and B layouts)

  // ---- init LDS ----
  if (tid < ROWS) {
    xh[0][tid] = (_Float16)0.0f;
    xh[1][tid] = (_Float16)0.0f;
  }
  if (tid < NPART) {
    partials[0][tid] = 0.0f;
    partials[1][tid] = 0.0f;
  }
  if (tid == 0) xh[0][U_SLOT] = (_Float16)u[0];   // u-channel for step 0

  // ---- preload augmented weights [W | w_in] into A-operand registers ----
  // A 16x32 f16 layout (ISA 7.12.2): lane l<16 -> M=l,   K = 32c + (h<8 ? h : h+8)
  //                                  lane l>=16 -> M=l-16, same with K += 8
  v16h a[NCHUNK];
  {
    const int m = 16 * wv + (lane & 15);  // this wave's tile rows
#pragma unroll
    for (int c = 0; c < NCHUNK; ++c) {
      v16h f;
#pragma unroll
      for (int h = 0; h < 16; ++h) {
        const int k = 32 * c + g + (h < 8 ? h : h + 8);
        float w = 0.0f;
        if (m < HIDDEN) {
          if (k < HIDDEN)       w = w_res[m * HIDDEN + k];
          else if (k == U_SLOT) w = w_in[m];     // augmented input column
        }
        f[h] = (_Float16)w;
      }
      a[c] = f;
    }
  }

  // D rows owned by this lane: R_r = 16w + g + r  (r = 0..7)
  float wo[8];
#pragma unroll
  for (int r = 0; r < 8; ++r) {
    const int R = 16 * wv + g + r;
    wo[r] = (R < HIDDEN) ? w_out[R] : 0.0f;
  }

  const bool is_writer = (lane == 0) || (lane == 16);   // one per lane group
  const int  pslot     = 2 * wv + (lane >> 4);          // partials slot
  const bool is_upatch = (wv == 12) && (lane == 16);    // owns row 200
  const bool is_drain  = (wv == 0) && (lane == 1);

  __syncthreads();

  for (int t = 0; t < T_STEPS; ++t) {
    // uniform scalar prefetch of next input (consumed by the u-channel patch)
    const float unext = u[(t + 1 < T_STEPS) ? (t + 1) : t];

    const _Float16* __restrict__ xr = xh[t & 1];        // [x_{t-1}; u_t]
    _Float16* __restrict__       xw = xh[(t & 1) ^ 1];  // [x_t; u_{t+1}]

    // ---- preload ALL B fragments (14 ds_load_b128, broadcast) ----
    // B mirror layout: halves 0..7 <- x[32c+g .. +7], 8..15 <- x[32c+g+16 .. +23]
    v16h b[NCHUNK];
#pragma unroll
    for (int c = 0; c < NCHUNK; ++c) {
      const v8h lo = *(const v8h*)(&xr[32 * c + g]);
      const v8h hi = *(const v8h*)(&xr[32 * c + g + 16]);
      b[c] = __builtin_shufflevector(lo, hi,
          0, 1, 2, 3, 4, 5, 6, 7, 8, 9, 10, 11, 12, 13, 14, 15);
    }

    // ---- y = W x + w_in u_t : single 7-deep WMMA accumulation chain ----
    v8f cc = {0.f, 0.f, 0.f, 0.f, 0.f, 0.f, 0.f, 0.f};
#pragma unroll
    for (int c = 0; c < NCHUNK; ++c) {
      cc = __builtin_amdgcn_wmma_f32_16x16x32_f16(false, a[c], false, b[c],
                                                  (short)0, cc, false, false);
    }

    // ---- x_t = tanh(y); readout partial (pairwise tree, no cross-lane op) ----
    float tv[8];
    v8h nv;
#pragma unroll
    for (int r = 0; r < 8; ++r) {
      const float v = fast_tanh(cc[r]);
      tv[r] = v;
      nv[r] = (_Float16)v;
    }
    const float part =
        ((tv[0] * wo[0] + tv[1] * wo[1]) + (tv[2] * wo[2] + tv[3] * wo[3])) +
        ((tv[4] * wo[4] + tv[5] * wo[5]) + (tv[6] * wo[6] + tv[7] * wo[7]));

    // D is replicated across N; one lane per group writes its 8 rows + partial.
    if (is_writer) {
      *(v8h*)(&xw[16 * wv + g]) = nv;          // rows 16w+g .. 16w+g+7
      partials[t & 1][pslot]    = part;        // plain store, parallel banks
    }
    // u-channel patch: row 200 is in wave 12 / lane 16's block (DS is in-order
    // within a wave, so this overwrites the padded tanh(0) value).
    if (is_upatch) {
      xw[U_SLOT] = (_Float16)unext;
    }

    // ---- split barrier: signal, drain previous step's readout, then wait ----
    asm volatile("s_wait_dscnt 0x0" ::: "memory");   // state/partials stores done
    asm volatile("s_barrier_signal -1" ::: "memory");
    if (is_drain && t > 0) {                         // hidden between signal/wait
      const float4* p4 = (const float4*)partials[(t & 1) ^ 1];
      float s = 0.0f;
#pragma unroll
      for (int i = 0; i < NPART / 4; ++i) {
        const float4 q = p4[i];
        s += (q.x + q.y) + (q.z + q.w);
      }
      if (t - 1 >= WASHOUT) out[t - 1 - WASHOUT] = s;
    }
    asm volatile("s_barrier_wait -1" ::: "memory");
  }

  // drain final step
  if (is_drain) {
    const float4* p4 = (const float4*)partials[(T_STEPS - 1) & 1];
    float s = 0.0f;
#pragma unroll
    for (int i = 0; i < NPART / 4; ++i) {
      const float4 q = p4[i];
      s += (q.x + q.y) + (q.z + q.w);
    }
    out[T_STEPS - 1 - WASHOUT] = s;
  }
}

extern "C" void kernel_launch(void* const* d_in, const int* in_sizes, int n_in,
                              void* d_out, int out_size, void* d_ws, size_t ws_size,
                              hipStream_t stream) {
  const float* u     = (const float*)d_in[0];
  const float* w_res = (const float*)d_in[1];
  const float* w_in  = (const float*)d_in[2];
  const float* w_out = (const float*)d_in[3];
  float* out = (float*)d_out;
  (void)in_sizes; (void)n_in; (void)out_size; (void)d_ws; (void)ws_size;

  esn_scan_kernel<<<dim3(1), dim3(512), 0, stream>>>(u, w_res, w_in, w_out, out);
}
